// BEVFusion_15384572854352
// MI455X (gfx1250) — matrix operand order repaired and checked by reading, not compiled, not run on hardware
//
#include <hip/hip_runtime.h>
#include <hip/hip_bf16.h>
#include <math.h>

typedef __attribute__((ext_vector_type(16))) _Float16 v16h;
typedef __attribute__((ext_vector_type(8)))  _Float16 v8h;
typedef __attribute__((ext_vector_type(8)))  float    v8f;
typedef __attribute__((ext_vector_type(4)))  int      v4i;

#if defined(__HIP_DEVICE_COMPILE__) && \
    __has_builtin(__builtin_amdgcn_global_load_async_to_lds_b128) && \
    __has_builtin(__builtin_amdgcn_s_wait_asynccnt)
#define HAVE_ASYNC_LDS 1
#else
#define HAVE_ASYNC_LDS 0
#endif

__device__ __forceinline__ v8f wmma16x16x32(v16h a, v16h b, v8f c) {
  // D = A(16x32 f16) * B(32x16 f16) + C(16x16 f32)
  return __builtin_amdgcn_wmma_f32_16x16x32_f16(false, a, false, b, (short)0, c, false, false);
}

// convert 16 contiguous f32 -> v16h via four float4 loads
__device__ __forceinline__ v16h cvt16(const float* __restrict__ p) {
  v16h r;
  #pragma unroll
  for (int i = 0; i < 4; i++) {
    float4 t = ((const float4*)p)[i];
    r[4*i]   = (_Float16)t.x;
    r[4*i+1] = (_Float16)t.y;
    r[4*i+2] = (_Float16)t.z;
    r[4*i+3] = (_Float16)t.w;
  }
  return r;
}

// =====================================================================
// Generic tiled WMMA GEMM (layouts as compile-time template params).
//  C[M,N] row-major (batched via blockIdx.z) = epi(A @ B)
//  TA==0: A[M,K] row-major ; TA==1: A stored [K,M] (channel-first acts)
//  TB==0: B stored [K,N]   ; TB==1: B stored [N,K] (conv OIHW 1x1 weights)
//  K and N are multiples of 32 for every call in this network; only M is
//  ragged (guarded by address-clamp; garbage rows >= M never reach memory).
//  epi: y = acc*s + t (+bias); s,t from eval BN(g,b,m,v) if g!=null; opt ReLU.
// Block = 128 threads = 4 waves; 32x32 macro-tile; K staged via LDS as f16.
// LDS rows padded to 40 halves (80B) so fragment reads are 16B ds_load_b128.
// B tile stored transposed (Bs[n][k]) so B-fragments are contiguous in k.
// =====================================================================
template<int TA, int TB>
__global__ void gemm_wmma(const float* __restrict__ A, const float* __restrict__ Bw,
                          float* __restrict__ C, int M, int N, int K,
                          long aStride, long cStride,
                          const float* __restrict__ bias,
                          const float* __restrict__ bng, const float* __restrict__ bnb,
                          const float* __restrict__ bnm, const float* __restrict__ bnv,
                          int relu)
{
  __shared__ __align__(16) _Float16 As[32][40];   // As[m][k]
  __shared__ __align__(16) _Float16 Bs[32][40];   // Bs[n][k]
  const int bz = blockIdx.z;
  const float* Ab = A + (long)bz * aStride;
  float* Cb = C + (long)bz * cStride;
  const int tid  = threadIdx.x;
  const int lane = tid & 31;
  const int wid  = tid >> 5;
  const int wm = wid >> 1, wn = wid & 1;
  const int mBlk = blockIdx.x * 32;
  const int nBlk = blockIdx.y * 32;
  v8f acc = {};
  for (int kt = 0; kt < K; kt += 32) {
    const int ktn = (kt + 32 < K) ? (kt + 32) : kt;   // next-tile prefetch base
    // ---- stage A tile (1024 elems = 128 thr * 2 reps * float4) ----
    #pragma unroll
    for (int rep = 0; rep < 2; rep++) {
      int linear = rep * 512 + tid * 4;
      if (TA == 0) {
        int mm = linear >> 5, kk = linear & 31;      // 4 consecutive k
        int gm = mBlk + mm;
        int gmc = (gm < M) ? gm : (M - 1);
        float msk = (gm < M) ? 1.f : 0.f;
        float4 t = *(const float4*)(Ab + (size_t)gmc * K + (kt + kk));
        __builtin_prefetch(Ab + (size_t)gmc * K + (ktn + kk), 0, 3);
        As[mm][kk]     = (_Float16)(t.x * msk);
        As[mm][kk + 1] = (_Float16)(t.y * msk);
        As[mm][kk + 2] = (_Float16)(t.z * msk);
        As[mm][kk + 3] = (_Float16)(t.w * msk);
      } else {
        int kk = linear >> 5, mm = linear & 31;      // 4 consecutive m
        int gm = mBlk + mm;
        const float* src = Ab + (size_t)(kt + kk) * M;
        __builtin_prefetch(Ab + (size_t)(ktn + kk) * M + ((gm < M) ? gm : 0), 0, 3);
        if (gm + 3 < M) {
          float4 t = *(const float4*)(src + gm);
          As[mm][kk]     = (_Float16)t.x;
          As[mm + 1][kk] = (_Float16)t.y;
          As[mm + 2][kk] = (_Float16)t.z;
          As[mm + 3][kk] = (_Float16)t.w;
        } else {
          #pragma unroll
          for (int j = 0; j < 4; j++) {
            int g = gm + j;
            int gc = (g < M) ? g : (M - 1);
            float msk = (g < M) ? 1.f : 0.f;
            As[mm + j][kk] = (_Float16)(src[gc] * msk);
          }
        }
      }
      // ---- stage B tile (always fully in range: K%32==0, N%32==0) ----
      if (TB == 0) {
        int kk = linear >> 5, nn = linear & 31;      // 4 consecutive n
        float4 t = *(const float4*)(Bw + (size_t)(kt + kk) * N + (nBlk + nn));
        __builtin_prefetch(Bw + (size_t)(ktn + kk) * N + (nBlk + nn), 0, 3);
        Bs[nn][kk]     = (_Float16)t.x;
        Bs[nn + 1][kk] = (_Float16)t.y;
        Bs[nn + 2][kk] = (_Float16)t.z;
        Bs[nn + 3][kk] = (_Float16)t.w;
      } else {
        int nn = linear >> 5, kk = linear & 31;      // 4 consecutive k
        float4 t = *(const float4*)(Bw + (size_t)(nBlk + nn) * K + (kt + kk));
        __builtin_prefetch(Bw + (size_t)(nBlk + nn) * K + (ktn + kk), 0, 3);
        Bs[nn][kk]     = (_Float16)t.x;
        Bs[nn][kk + 1] = (_Float16)t.y;
        Bs[nn][kk + 2] = (_Float16)t.z;
        Bs[nn][kk + 3] = (_Float16)t.w;
      }
    }
    __syncthreads();
    // ---- fragments: 16B LDS vector reads ----
    const _Float16* arow = &As[wm * 16 + (lane & 15)][0];
    const _Float16* brow = &Bs[wn * 16 + (lane & 15)][0];
    const int akb = (lane >= 16) ? 8 : 0;
    const int bkb = (lane >= 16) ? 8 : 0;
    v8h alo = *(const v8h*)(arow + akb);
    v8h ahi = *(const v8h*)(arow + akb + 16);
    v8h blo = *(const v8h*)(brow + bkb);
    v8h bhi = *(const v8h*)(brow + bkb + 16);
    v16h af, bf;
    #pragma unroll
    for (int i = 0; i < 8; i++) {
      af[i] = alo[i]; af[8 + i] = ahi[i];
      bf[i] = blo[i]; bf[8 + i] = bhi[i];
    }
    acc = wmma16x16x32(af, bf, acc);
    __syncthreads();
  }
  const int gn = nBlk + wn * 16 + (lane & 15);
  {
    float s = 1.f, t = 0.f;
    if (bng) { s = bng[gn] * rsqrtf(bnv[gn] + 1e-5f); t = bnb[gn] - bnm[gn] * s; }
    if (bias) t += bias[gn];
    const int mBase = mBlk + wm * 16 + ((lane >= 16) ? 8 : 0);
    #pragma unroll
    for (int r = 0; r < 8; r++) {
      int gm = mBase + r;
      if (gm < M) {
        float y = acc[r] * s + t;
        if (relu) y = fmaxf(y, 0.f);
        Cb[(size_t)gm * N + gn] = y;
      }
    }
  }
}

// =====================================================================
// Flash cross-attention. q,k,v,o: [B, 4096, 128] channel-last (head h uses
// dims h*32..h*32+31). Grid: (B*NH, N/64); 128 threads = 4 waves; each wave
// owns 16 query rows; streams 32-key tiles; online softmax.
// V tiles staged through LDS via async global->LDS copies (double-buffered)
// when the toolchain exposes the gfx1250 async builtins.
// =====================================================================
__global__ void flash_attn(const float* __restrict__ q, const float* __restrict__ k,
                           const float* __restrict__ v, float* __restrict__ o)
{
  const int N = 4096;
  const int bh = blockIdx.x;
  const int b = bh >> 2, h = bh & 3;
  const int tid = threadIdx.x;
  const int lane = tid & 31;
  const int wv = tid >> 5;
  const int q0 = blockIdx.y * 64 + wv * 16;
  const float* qb = q + (long)b * N * 128 + h * 32;
  const float* kb = k + (long)b * N * 128 + h * 32;
  const float* vb = v + (long)b * N * 128 + h * 32;
  __shared__ __align__(16) _Float16 Pst[4][16][40];
#if HAVE_ASYNC_LDS
  __shared__ __align__(16) float Vs[2][32][36];   // double-buffered V tile
#endif

  v16h qf;
  {
    int m = lane & 15, kbase = (lane >= 16) ? 16 : 0;
    const float* qr = qb + (long)(q0 + m) * 128;
    // two contiguous 8-float runs: dims [kbase..kbase+7], [8+kbase..15+kbase]
    #pragma unroll
    for (int i = 0; i < 2; i++) {
      float4 t0 = ((const float4*)(qr + kbase))[i];
      float4 t1 = ((const float4*)(qr + 8 + kbase))[i];
      qf[4*i]   = (_Float16)t0.x; qf[4*i+1] = (_Float16)t0.y;
      qf[4*i+2] = (_Float16)t0.z; qf[4*i+3] = (_Float16)t0.w;
      qf[8+4*i]   = (_Float16)t1.x; qf[8+4*i+1] = (_Float16)t1.y;
      qf[8+4*i+2] = (_Float16)t1.z; qf[8+4*i+3] = (_Float16)t1.w;
    }
  }
  v8f acc0 = {}, acc1 = {};
  float rm[8], rl[8];
  #pragma unroll
  for (int r = 0; r < 8; r++) { rm[r] = -1e30f; rl[r] = 0.f; }
  const float sc = 0.17677669529663687f;   // 1/sqrt(32)

  for (int j0 = 0; j0 < N; j0 += 32) {
    const int n = lane & 15;
    const int kb2 = (lane >= 16) ? 16 : 0;
#if HAVE_ASYNC_LDS
    const int buf = (j0 >> 5) & 1;
    // kick off async V-tile copy (32 keys x 32 dims f32, 16B chunks)
    #pragma unroll
    for (int rep = 0; rep < 2; rep++) {
      int chunk = rep * 128 + tid;            // 0..255
      int r = chunk >> 3, c4 = (chunk & 7) * 4;
      const float* gp = vb + (size_t)(j0 + r) * 128 + c4;
      __builtin_amdgcn_global_load_async_to_lds_b128(
          (__attribute__((address_space(1))) v4i*)gp,
          (__attribute__((address_space(3))) v4i*)&Vs[buf][r][c4], 0, 0);
    }
#endif
    // K fragments: lane holds one key row, 16 contiguous dims
    v16h kf0 = cvt16(kb + (long)(j0 + n) * 128 + kb2);
    v16h kf1 = cvt16(kb + (long)(j0 + 16 + n) * 128 + kb2);
    {   // prefetch next key tile
      int jn = (j0 + 32 < N) ? (j0 + 32) : j0;
      __builtin_prefetch(kb + (long)(jn + n) * 128 + kb2, 0, 3);
      __builtin_prefetch(kb + (long)(jn + 16 + n) * 128 + kb2, 0, 3);
    }
    v8f s0 = {}, s1 = {};
    s0 = wmma16x16x32(qf, kf0, s0);   // keys j0..j0+15
    s1 = wmma16x16x32(qf, kf1, s1);   // keys j0+16..j0+31
    float p0[8], p1[8];
    #pragma unroll
    for (int r = 0; r < 8; r++) {
      float a0 = s0[r] * sc, a1 = s1[r] * sc;
      float mx = fmaxf(a0, a1);
      #pragma unroll
      for (int ofs = 1; ofs < 16; ofs <<= 1)
        mx = fmaxf(mx, __shfl_xor(mx, ofs, 16));
      float nm = fmaxf(rm[r], mx);
      float al = __expf(rm[r] - nm);
      float e0 = __expf(a0 - nm), e1 = __expf(a1 - nm);
      float rs = e0 + e1;
      #pragma unroll
      for (int ofs = 1; ofs < 16; ofs <<= 1)
        rs += __shfl_xor(rs, ofs, 16);
      rl[r] = rl[r] * al + rs;
      rm[r] = nm;
      acc0[r] *= al;
      acc1[r] *= al;
      p0[r] = e0; p1[r] = e1;
    }
    __syncthreads();            // previous iter's Pst reads done
    {
      int mof = (lane >= 16) ? 8 : 0;   // C-frag -> LDS
      #pragma unroll
      for (int r = 0; r < 8; r++) {
        Pst[wv][r + mof][n]      = (_Float16)p0[r];
        Pst[wv][r + mof][16 + n] = (_Float16)p1[r];
      }
    }
#if HAVE_ASYNC_LDS
    __builtin_amdgcn_s_wait_asynccnt(0);   // own V-tile chunks landed in LDS
#endif
    __syncthreads();                       // all waves' Pst + V writes visible
    v16h pf;
    {
      int m = lane & 15, kbase = (lane >= 16) ? 16 : 0;  // LDS -> A-frag
      const _Float16* prow = &Pst[wv][m][0];
      v8h plo = *(const v8h*)(prow + kbase);
      v8h phi = *(const v8h*)(prow + kbase + 8);
      #pragma unroll
      for (int i = 0; i < 8; i++) { pf[i] = plo[i]; pf[8 + i] = phi[i]; }
    }
    v16h vf0, vf1;
    {
      #pragma unroll
      for (int i = 0; i < 8; i++) {
#if HAVE_ASYNC_LDS
        vf0[2*i]   = (_Float16)Vs[buf][kb2 + 2*i][n];
        vf0[2*i+1] = (_Float16)Vs[buf][kb2 + 2*i + 1][n];
        vf1[2*i]   = (_Float16)Vs[buf][kb2 + 2*i][16 + n];
        vf1[2*i+1] = (_Float16)Vs[buf][kb2 + 2*i + 1][16 + n];
#else
        const float* vr = vb + (long)(j0 + kb2 + 2*i) * 128;
        vf0[2*i]   = (_Float16)vr[n];
        vf0[2*i+1] = (_Float16)vr[128 + n];
        vf1[2*i]   = (_Float16)vr[16 + n];
        vf1[2*i+1] = (_Float16)vr[128 + 16 + n];
#endif
      }
    }
    acc0 = wmma16x16x32(pf, vf0, acc0);
    acc1 = wmma16x16x32(pf, vf1, acc1);
  }
  {
    int n = lane & 15, mof = (lane >= 16) ? 8 : 0;
    #pragma unroll
    for (int r = 0; r < 8; r++) {
      float invl = 1.f / rl[r];
      float* orow = o + ((long)b * N + q0 + r + mof) * 128 + h * 32;
      orow[n]      = acc0[r] * invl;
      orow[16 + n] = acc1[r] * invl;
    }
  }
}

// ===================== small direct kernels ==========================

__global__ void zero_f32(float* __restrict__ p, long n) {
  long i = (long)blockIdx.x * blockDim.x + threadIdx.x;
  if (i < n) p[i] = 0.f;
}

// grouped 3x3 conv, groups=8: in/out [B,HW,128] channel-last, w [128,16,3,3]
__global__ void grouped_conv3x3(const float* __restrict__ in, const float* __restrict__ w,
                                float* __restrict__ out, int H, int W,
                                const float* __restrict__ bng, const float* __restrict__ bnb,
                                const float* __restrict__ bnm, const float* __restrict__ bnv)
{
  int idx = blockIdx.x * blockDim.x + threadIdx.x;
  int HW = H * W;
  int co = idx & 127, p = idx >> 7;
  if (p >= HW) return;
  int b = blockIdx.z;
  int py = p / W, px = p % W;
  int g = co >> 4;
  float acc = 0.f;
  for (int dy = -1; dy <= 1; dy++)
    for (int dx = -1; dx <= 1; dx++) {
      int yy = py + dy, xx = px + dx;
      if (yy < 0 || yy >= H || xx < 0 || xx >= W) continue;
      const float* ip = in + ((long)b * HW + yy * W + xx) * 128 + g * 16;
      const float* wp = w + (long)co * 144 + (dy + 1) * 3 + (dx + 1);
      #pragma unroll
      for (int ci = 0; ci < 16; ci++) acc += ip[ci] * wp[ci * 9];
    }
  float s = bng[co] * rsqrtf(bnv[co] + 1e-5f);
  float t = bnb[co] - bnm[co] * s;
  out[((long)b * HW + p) * 128 + co] = fmaxf(acc * s + t, 0.f);
}

// confidence 3x3 conv: cat(dl[B,HW,64], red[B,HW,128]) -> [B,HW,64]; w [64,192,3,3]
__global__ void cf_conv3x3(const float* __restrict__ dl, const float* __restrict__ red,
                           const float* __restrict__ w, float* __restrict__ out,
                           int H, int W,
                           const float* __restrict__ bng, const float* __restrict__ bnb,
                           const float* __restrict__ bnm, const float* __restrict__ bnv)
{
  int idx = blockIdx.x * blockDim.x + threadIdx.x;
  int HW = H * W;
  int co = idx & 63, p = idx >> 6;
  if (p >= HW) return;
  int b = blockIdx.z;
  int py = p / W, px = p % W;
  float acc = 0.f;
  for (int dy = -1; dy <= 1; dy++)
    for (int dx = -1; dx <= 1; dx++) {
      int yy = py + dy, xx = px + dx;
      if (yy < 0 || yy >= H || xx < 0 || xx >= W) continue;
      long pp = (long)b * HW + yy * W + xx;
      const float* wp = w + (long)co * 192 * 9 + (dy + 1) * 3 + (dx + 1);
      const float* dp = dl + pp * 64;
      const float* rp = red + pp * 128;
      for (int ci = 0; ci < 64; ci++) acc += dp[ci] * wp[ci * 9];
      for (int ci = 0; ci < 128; ci++) acc += rp[ci] * wp[(64 + ci) * 9];
    }
  float s = bng[co] * rsqrtf(bnv[co] + 1e-5f);
  float t = bnb[co] - bnm[co] * s;
  out[((long)b * HW + p) * 64 + co] = fmaxf(acc * s + t, 0.f);
}

// depth softmax + bins + pinhole lift geometry -> y_idx, valid
__global__ void depth_geom(const float* __restrict__ dl, const float* __restrict__ Kmat,
                           const float* __restrict__ Tmat, int* __restrict__ yidx,
                           int* __restrict__ validb, int H, int W)
{
  int p = blockIdx.x * blockDim.x + threadIdx.x;
  int HW = H * W;
  if (p >= HW) return;
  int b = blockIdx.z;
  const float* d = dl + ((long)b * HW + p) * 64;
  float mx = -1e30f;
  for (int c = 0; c < 64; c++) mx = fmaxf(mx, d[c] * 10.f);
  float se = 0.f, sd = 0.f;
  const float lb = logf(35.0f) / 63.0f;
  for (int c = 0; c < 64; c++) {
    float e = __expf(d[c] * 10.f - mx);
    se += e;
    sd += e * __expf((float)c * lb);
  }
  float depth = sd / se;
  const float* Km = Kmat + (long)b * 9;
  float k00 = Km[0], k01 = Km[1], k02 = Km[2];
  float k10 = Km[3], k11 = Km[4], k12 = Km[5];
  float k20 = Km[6], k21 = Km[7], k22 = Km[8];
  float A0 = k11 * k22 - k12 * k21, A1 = k02 * k21 - k01 * k22, A2 = k01 * k12 - k02 * k11;
  float A3 = k12 * k20 - k10 * k22, A4 = k00 * k22 - k02 * k20, A5 = k02 * k10 - k00 * k12;
  float A6 = k10 * k21 - k11 * k20, A7 = k01 * k20 - k00 * k21, A8 = k00 * k11 - k01 * k10;
  float inv = 1.f / (k00 * A0 + k01 * A3 + k02 * A6);
  float u = (float)(p % W), vv = (float)(p / W);
  float X = (A0 * u + A1 * vv + A2) * inv;
  float Y = (A3 * u + A4 * vv + A5) * inv;
  float Z = (A6 * u + A7 * vv + A8) * inv;
  float cx = depth * X, cy = depth * Y, cz = depth * Z;
  const float* T = Tmat + (long)b * 16;
  float ex = T[0] * cx + T[1] * cy + T[2] * cz + T[3];
  float ey = T[4] * cx + T[5] * cy + T[6] * cz + T[7];
  int bx = (int)(ex / 0.8f + 32.0f);
  int by = (int)(ey / 0.8f + 32.0f);
  int ok = (bx >= 0 && bx < 64 && by >= 0 && by < 64);
  validb[(long)b * HW + p] = ok;
  yidx[(long)b * HW + p] = ok ? by : 0;
}

__global__ void conf_head(const float* __restrict__ cf1, const float* __restrict__ w2,
                          const float* __restrict__ b2, float* __restrict__ conf, int HW)
{
  int p = blockIdx.x * blockDim.x + threadIdx.x;
  if (p >= HW) return;
  int b = blockIdx.z;
  const float* cp = cf1 + ((long)b * HW + p) * 64;
  float acc = b2[0];
  for (int c = 0; c < 64; c++) acc += cp[c] * w2[c];
  conf[(long)b * HW + p] = 1.f / (1.f + __expf(-acc));
}

// deterministic BEV splat (gather form of scatter_add over image rows):
// concat[b, chOff+c, y*64 + wcol] = sum_i red[b, i*W+wcol, c]*conf*valid[y==yidx]
__global__ void gather_bev(const float* __restrict__ red, const float* __restrict__ conf,
                           const int* __restrict__ yidx, const int* __restrict__ validb,
                           float* __restrict__ concat, int H, int W, int chOff)
{
  int idx = blockIdx.x * blockDim.x + threadIdx.x;
  int HW = H * W;
  int wcol = idx % W;
  int y = (idx / W) & 63;
  int c = idx / (W * 64);
  if (c >= 128) return;
  int b = blockIdx.z;
  float acc = 0.f;
  for (int i = 0; i < H; i++) {
    int p = i * W + wcol;
    long bp = (long)b * HW + p;
    if (validb[bp] && yidx[bp] == y)
      acc += red[bp * 128 + c] * conf[bp];
  }
  concat[((long)b * 384 + chOff + c) * 4096 + y * 64 + wcol] = acc;
}

// x [B,4096,128] += pos_embed[1,128,64,64]
__global__ void add_pos(float* __restrict__ x, const float* __restrict__ pos)
{
  int idx = blockIdx.x * blockDim.x + threadIdx.x;
  int c = idx & 127, p = (idx >> 7) & 4095;
  int b = blockIdx.z;
  x[((long)b * 4096 + p) * 128 + c] += pos[(long)c * 4096 + p];
}

// residual + LayerNorm; qres==1 replays the reference's head-scrambled reshape
__global__ void residual_ln(const float* __restrict__ xa, const float* __restrict__ xb,
                            const float* __restrict__ qsrc,
                            const float* __restrict__ g, const float* __restrict__ beta,
                            float* __restrict__ out, int qres)
{
  int row = blockIdx.x;            // b*4096 + n
  int c = threadIdx.x;             // 0..127
  int b = row >> 12, n = row & 4095;
  float v1;
  if (qres) {
    long lin = (long)n * 128 + c;
    int h = (int)(lin >> 17);                 // /(4096*32)
    long rem = lin & ((1L << 17) - 1);
    int ns = (int)(rem >> 5), d = (int)(rem & 31);
    v1 = qsrc[((long)b * 4096 + ns) * 128 + h * 32 + d];
  } else {
    v1 = xa[(long)row * 128 + c];
  }
  float v2 = xb[(long)row * 128 + c];
  float x = v1 + v2;
  __shared__ float sbuf[4];
  int lane = c & 31, wv = c >> 5;
  float s = x;
  #pragma unroll
  for (int o = 16; o > 0; o >>= 1) s += __shfl_xor(s, o, 32);
  if (lane == 0) sbuf[wv] = s;
  __syncthreads();
  float mean = (sbuf[0] + sbuf[1] + sbuf[2] + sbuf[3]) * (1.f / 128.f);
  float dx = x - mean;
  float s2 = dx * dx;
  #pragma unroll
  for (int o = 16; o > 0; o >>= 1) s2 += __shfl_xor(s2, o, 32);
  __syncthreads();
  if (lane == 0) sbuf[wv] = s2;
  __syncthreads();
  float var = (sbuf[0] + sbuf[1] + sbuf[2] + sbuf[3]) * (1.f / 128.f);
  out[(long)row * 128 + c] = dx * rsqrtf(var + 1e-5f) * g[c] + beta[c];
}

// [B,4096,128] -> [B,128,4096]
__global__ void to_nchw(const float* __restrict__ in, float* __restrict__ out)
{
  int idx = blockIdx.x * blockDim.x + threadIdx.x;
  int c = idx & 127, p = (idx >> 7) & 4095;
  int b = blockIdx.z;
  out[((long)b * 128 + c) * 4096 + p] = in[((long)b * 4096 + p) * 128 + c];
}

// =====================================================================
// Host side. Input flattening assumption: the 'params' pytree is flattened
// with dict keys sorted at every level (jax tree order), followed by the
// remaining setup_inputs entries in insertion order.
// =====================================================================
enum {
  IDX_F1_B = 0, IDX_F1_W, IDX_F2_B, IDX_F2_W, IDX_K_B, IDX_K_W,
  IDX_LN1_B, IDX_LN1_G, IDX_LN2_B, IDX_LN2_G, IDX_O_B, IDX_O_W,
  IDX_Q_B, IDX_Q_W, IDX_V_B, IDX_V_W,                 // attn: 0..15
  IDX_LIFT0 = 16,                                      // 3 stages * 24 leaves
  IDX_POS = 88,
  IDX_IBN_B = 89, IDX_IBN_G, IDX_IBN_M, IDX_IBN_V, IDX_I_W,
  IDX_LBN_B = 94, IDX_LBN_G, IDX_LBN_M, IDX_LBN_V, IDX_L_W,
  IDX_SF_B2 = 99, IDX_SF_BN_B, IDX_SF_BN_G, IDX_SF_BN_M, IDX_SF_BN_V,
  IDX_SF_W1, IDX_SF_W2,
  IDX_LIDAR = 106, IDX_F3, IDX_F4, IDX_F5, IDX_KMAT, IDX_TMAT
};
enum {
  L_CF_B2 = 0, L_CF_BN_B, L_CF_BN_G, L_CF_BN_M, L_CF_BN_V, L_CF_W1, L_CF_W2,
  L_DN_B2, L_DN_BN_B, L_DN_BN_G, L_DN_BN_M, L_DN_BN_V, L_DN_W1, L_DN_W2,
  L_FR_BN1_B, L_FR_BN1_G, L_FR_BN1_M, L_FR_BN1_V,
  L_FR_BN2_B, L_FR_BN2_G, L_FR_BN2_M, L_FR_BN2_V, L_FR_W1, L_FR_W2
};

extern "C" void kernel_launch(void* const* d_in, const int* in_sizes, int n_in,
                              void* d_out, int out_size, void* d_ws, size_t ws_size,
                              hipStream_t stream)
{
  (void)in_sizes; (void)n_in; (void)out_size; (void)ws_size;
  const int B = 2;
  auto F = [&](int i) { return (const float*)d_in[i]; };

  // ---- workspace (bump allocator over d_ws) ----
  float* ws = (float*)d_ws;
  size_t off = 0;
  auto alloc = [&](size_t n) { float* p = ws + off; off += n; return p; };
  float* concat = alloc((size_t)B * 384 * 4096);   // BEV concat [B,384,64,64]
  float* sfm    = alloc((size_t)B * 4096 * 256);
  float* imgbev = alloc((size_t)B * 4096 * 128);
  float* lid    = alloc((size_t)B * 4096 * 128);
  float* img    = alloc((size_t)B * 4096 * 128);
  float* qbuf   = alloc((size_t)B * 4096 * 128);
  float* kbuf   = alloc((size_t)B * 4096 * 128);
  float* vbuf   = alloc((size_t)B * 4096 * 128);
  float* attn   = alloc((size_t)B * 4096 * 128);
  float* out1   = alloc((size_t)B * 4096 * 128);
  const int MAXHW = 704;                           // stage3 = 16*44
  float* red1  = alloc((size_t)B * MAXHW * 128);
  float* red   = alloc((size_t)B * MAXHW * 128);
  float* dl1   = alloc((size_t)B * MAXHW * 128);
  float* dl    = alloc((size_t)B * MAXHW * 64);
  float* cf1   = alloc((size_t)B * MAXHW * 64);
  float* confb = alloc((size_t)B * MAXHW);
  int* yidx    = (int*)alloc((size_t)B * MAXHW);
  int* validb  = (int*)alloc((size_t)B * MAXHW);
  // aliases over dead regions:
  float* ffnh  = concat;   // needs B*4096*512 = 4,194,304 <= concat+sfm region
  float* oproj = imgbev;   // imgbev dead after proj-img GEMM
  float* ffno  = attn;     // attn dead after O projection
  float* out2  = lid;      // lid dead after Q projection

  auto gemm = [&](const float* A, const float* Bw, float* C, int M, int N, int K,
                  int tA, int tB, long aS, long cS, const float* bias,
                  const float* g, const float* bb, const float* m, const float* vv,
                  int relu) {
    dim3 gr((M + 31) / 32, (N + 31) / 32, B);
    if (tA == 0 && tB == 0)
      gemm_wmma<0,0><<<gr, 128, 0, stream>>>(A, Bw, C, M, N, K, aS, cS, bias, g, bb, m, vv, relu);
    else if (tA == 0 && tB == 1)
      gemm_wmma<0,1><<<gr, 128, 0, stream>>>(A, Bw, C, M, N, K, aS, cS, bias, g, bb, m, vv, relu);
    else if (tA == 1 && tB == 1)
      gemm_wmma<1,1><<<gr, 128, 0, stream>>>(A, Bw, C, M, N, K, aS, cS, bias, g, bb, m, vv, relu);
    else
      gemm_wmma<1,0><<<gr, 128, 0, stream>>>(A, Bw, C, M, N, K, aS, cS, bias, g, bb, m, vv, relu);
  };

  // ---- zero the BEV concat buffer (ws is not re-poisoned between replays) ----
  {
    long n = (long)B * 384 * 4096;
    zero_f32<<<dim3((unsigned)((n + 255) / 256)), 256, 0, stream>>>(concat, n);
  }

  // ---- lift stages ----
  const int stage_cin[3] = {256, 384, 512};
  const int stage_H[3]   = {16, 8, 4};
  const int stage_W[3]   = {44, 22, 11};
  const int stage_feat[3] = {IDX_F3, IDX_F4, IDX_F5};
  for (int s = 0; s < 3; s++) {
    int base = IDX_LIFT0 + s * 24;
    int cin = stage_cin[s], H = stage_H[s], W = stage_W[s], HW = H * W;
    const float* feat = F(stage_feat[s]);
    // feature reduce 1x1 (+BN+ReLU)
    gemm(feat, F(base + L_FR_W1), red1, HW, 128, cin, 1, 1,
         (long)cin * HW, (long)HW * 128, nullptr,
         F(base + L_FR_BN1_G), F(base + L_FR_BN1_B), F(base + L_FR_BN1_M), F(base + L_FR_BN1_V), 1);
    // grouped 3x3 (+BN+ReLU)
    grouped_conv3x3<<<dim3((HW * 128 + 255) / 256, 1, B), 256, 0, stream>>>(
        red1, F(base + L_FR_W2), red, H, W,
        F(base + L_FR_BN2_G), F(base + L_FR_BN2_B), F(base + L_FR_BN2_M), F(base + L_FR_BN2_V));
    // depth head
    gemm(feat, F(base + L_DN_W1), dl1, HW, 128, cin, 1, 1,
         (long)cin * HW, (long)HW * 128, nullptr,
         F(base + L_DN_BN_G), F(base + L_DN_BN_B), F(base + L_DN_BN_M), F(base + L_DN_BN_V), 1);
    gemm(dl1, F(base + L_DN_W2), dl, HW, 64, 128, 0, 1,
         (long)HW * 128, (long)HW * 64, F(base + L_DN_B2),
         nullptr, nullptr, nullptr, nullptr, 0);
    depth_geom<<<dim3((HW + 127) / 128, 1, B), 128, 0, stream>>>(
        dl, F(IDX_KMAT), F(IDX_TMAT), yidx, validb, H, W);
    // confidence head
    cf_conv3x3<<<dim3((HW * 64 + 255) / 256, 1, B), 256, 0, stream>>>(
        dl, red, F(base + L_CF_W1), cf1, H, W,
        F(base + L_CF_BN_G), F(base + L_CF_BN_B), F(base + L_CF_BN_M), F(base + L_CF_BN_V));
    conf_head<<<dim3((HW + 127) / 128, 1, B), 128, 0, stream>>>(
        cf1, F(base + L_CF_W2), F(base + L_CF_B2), confb, HW);
    // deterministic BEV splat
    gather_bev<<<dim3((128 * 64 * W + 255) / 256, 1, B), 256, 0, stream>>>(
        red, confb, yidx, validb, concat, H, W, s * 128);
  }

  // ---- scale fusion ----
  gemm(concat, F(IDX_SF_W1), sfm, 4096, 256, 384, 1, 1,
       (long)384 * 4096, (long)4096 * 256, nullptr,
       F(IDX_SF_BN_G), F(IDX_SF_BN_B), F(IDX_SF_BN_M), F(IDX_SF_BN_V), 1);
  gemm(sfm, F(IDX_SF_W2), imgbev, 4096, 128, 256, 0, 1,
       (long)4096 * 256, (long)4096 * 128, F(IDX_SF_B2),
       nullptr, nullptr, nullptr, nullptr, 0);

  // ---- modality projections + positional embedding ----
  gemm(F(IDX_LIDAR), F(IDX_L_W), lid, 4096, 128, 128, 1, 1,
       (long)128 * 4096, (long)4096 * 128, nullptr,
       F(IDX_LBN_G), F(IDX_LBN_B), F(IDX_LBN_M), F(IDX_LBN_V), 1);
  gemm(imgbev, F(IDX_I_W), img, 4096, 128, 128, 0, 1,
       (long)4096 * 128, (long)4096 * 128, nullptr,
       F(IDX_IBN_G), F(IDX_IBN_B), F(IDX_IBN_M), F(IDX_IBN_V), 1);
  add_pos<<<dim3(2048, 1, B), 256, 0, stream>>>(lid, F(IDX_POS));
  add_pos<<<dim3(2048, 1, B), 256, 0, stream>>>(img, F(IDX_POS));

  // ---- cross attention ----
  const long nc = (long)4096 * 128;
  gemm(lid, F(IDX_Q_W), qbuf, 4096, 128, 128, 0, 0, nc, nc, F(IDX_Q_B),
       nullptr, nullptr, nullptr, nullptr, 0);
  gemm(img, F(IDX_K_W), kbuf, 4096, 128, 128, 0, 0, nc, nc, F(IDX_K_B),
       nullptr, nullptr, nullptr, nullptr, 0);
  gemm(img, F(IDX_V_W), vbuf, 4096, 128, 128, 0, 0, nc, nc, F(IDX_V_B),
       nullptr, nullptr, nullptr, nullptr, 0);
  flash_attn<<<dim3(B * 4, 64), 128, 0, stream>>>(qbuf, kbuf, vbuf, attn);
  gemm(attn, F(IDX_O_W), oproj, 4096, 128, 128, 0, 0, nc, nc, F(IDX_O_B),
       nullptr, nullptr, nullptr, nullptr, 0);
  residual_ln<<<dim3(B * 4096), 128, 0, stream>>>(
      nullptr, oproj, qbuf, F(IDX_LN1_G), F(IDX_LN1_B), out1, 1);

  // ---- FFN + LN2 ----
  gemm(out1, F(IDX_F1_W), ffnh, 4096, 512, 128, 0, 0,
       nc, (long)4096 * 512, F(IDX_F1_B), nullptr, nullptr, nullptr, nullptr, 1);
  gemm(ffnh, F(IDX_F2_W), ffno, 4096, 128, 512, 0, 0,
       (long)4096 * 512, nc, F(IDX_F2_B), nullptr, nullptr, nullptr, nullptr, 0);
  residual_ln<<<dim3(B * 4096), 128, 0, stream>>>(
      out1, ffno, nullptr, F(IDX_LN2_G), F(IDX_LN2_B), out2, 0);

  // ---- final layout [B,128,64,64] ----
  to_nchw<<<dim3(2048, 1, B), 256, 0, stream>>>(out2, (float*)d_out);
}
